// MultiHeadAttention_25744033972761
// MI455X (gfx1250) — matrix operand order repaired and checked by reading, not compiled, not run on hardware
//
#include <hip/hip_runtime.h>
#include <hip/hip_bf16.h>

// ---------------------------------------------------------------------------
// MI455X (gfx1250) fused multi-head attention, bf16 WMMA + f32 accumulate.
//   S=2048, D=1024, H=16, DK=64.  All GEMMs on v_wmma_f32_16x16x32_bf16.
//   GEMMs use 32x64 register blocking per wave: 8 WMMAs per K=32 chunk from
//   6 KB of loads (0.75 KB/WMMA) instead of 2 KB/WMMA with 16x16 tiles.
// ---------------------------------------------------------------------------

#define S_LEN 2048
#define D_MOD 1024
#define NHEAD 16
#define DK_SZ 64

typedef __attribute__((ext_vector_type(16))) __bf16       v16bf;
typedef __attribute__((ext_vector_type(8)))  float        v8f;
typedef __attribute__((ext_vector_type(4)))  unsigned int v4u;

union FragU { v4u u4[2]; v16bf bf; };

// A fragment (16x32 bf16, row-major source, row stride ld elements).
// Lane = row (lane&15); K pattern: [8*half .. 8*half+7] and [16+8*half .. +7].
__device__ __forceinline__ v16bf load_a_frag(const __bf16* base, int ld) {
  const int lane = threadIdx.x & 31;
  const int row  = lane & 15;
  const int koff = (lane >> 4) << 3;                 // 0 or 8
  const __bf16* p = base + row * ld + koff;
  FragU f;
  f.u4[0] = *(const v4u*)(p);                        // K = koff .. koff+7
  f.u4[1] = *(const v4u*)(p + 16);                   // K = 16+koff .. +7
  return f.bf;
}

// B fragment (32x16 bf16). Source stored "N-major": row n holds the K-axis
// contiguously (i.e. B^T row-major). Lane = column N (lane&15); K pattern:
// [8*half .. 8*half+7] and [16+8*half .. +7] (same as A).
__device__ __forceinline__ v16bf load_b_frag(const __bf16* base, int ld) {
  const int lane = threadIdx.x & 31;
  const int col  = lane & 15;
  const int koff = (lane >> 4) << 3;                 // 0 or 8
  const __bf16* p = base + col * ld + koff;
  FragU f;
  f.u4[0] = *(const v4u*)(p);
  f.u4[1] = *(const v4u*)(p + 16);
  return f.bf;
}

__device__ __forceinline__ v8f wmma_bf16(v16bf a, v16bf b, v8f c) {
  return __builtin_amdgcn_wmma_f32_16x16x32_bf16(
      /*neg_a=*/false, a, /*neg_b=*/false, b,
      /*c_mod=*/(short)0, c, /*reuse_a=*/false, /*reuse_b=*/false);
}

// ---------------------------------------------------------------------------
// Precision conversion / layout kernels
// ---------------------------------------------------------------------------
__global__ void cvt_f32_bf16(const float* __restrict__ src,
                             __bf16* __restrict__ dst, int n) {
  int i = blockIdx.x * blockDim.x + threadIdx.x;
  if (i < n) dst[i] = (__bf16)src[i];
}

// Wq/Wk/Wv: [H, D, DK] f32 -> [H, DK, D] bf16 (B^T layout for WMMA B frags)
__global__ void cvt_transpose_w(const float* __restrict__ src,
                                __bf16* __restrict__ dst) {
  int i = blockIdx.x * blockDim.x + threadIdx.x;     // over H*D*DK = 1M
  int h = i >> 16;                                   // D*DK = 65536
  int r = i & 65535;
  int d = r >> 6;                                    // DK = 64
  int k = r & 63;
  dst[(h << 16) + (k << 10) + d] = (__bf16)src[i];
}

// Wo: [H*DK, D] f32 -> [D, H*DK] bf16
__global__ void cvt_transpose_wo(const float* __restrict__ src,
                                 __bf16* __restrict__ dst) {
  int i = blockIdx.x * blockDim.x + threadIdx.x;     // over 1M
  int m = i >> 10;
  int d = i & 1023;
  dst[(d << 10) + m] = (__bf16)src[i];
}

// ---------------------------------------------------------------------------
// Per-head projections: q/k -> [H,S,DK] bf16 ; v -> transposed vT [H,DK,S]
// One 32x64 output block per wave (2 row tiles x 4 col tiles), K=1024:
// per K=32 chunk: 2 A frags + 4 B frags -> 8 WMMAs.
// ---------------------------------------------------------------------------
__global__ void __launch_bounds__(256)
proj_kernel(const __bf16* __restrict__ Qb, const __bf16* __restrict__ Kb,
            const __bf16* __restrict__ Vb,
            const __bf16* __restrict__ WqT, const __bf16* __restrict__ WkT,
            const __bf16* __restrict__ WvT,
            const float* __restrict__ bq, const float* __restrict__ bk,
            const float* __restrict__ bv,
            __bf16* __restrict__ qh, __bf16* __restrict__ kh,
            __bf16* __restrict__ vT) {
  const int gw  = blockIdx.x * 8 + (threadIdx.x >> 5);   // 0..3071
  const int mat = gw >> 10;                              // 0=q 1=k 2=v
  const int rem = gw & 1023;
  const int h   = rem >> 6;                              // head
  const int st  = rem & 63;                              // 32-row tile

  const __bf16* X  = (mat == 0) ? Qb  : (mat == 1) ? Kb  : Vb;
  const __bf16* WT = (mat == 0) ? WqT : (mat == 1) ? WkT : WvT;
  const float*  bb = (mat == 0) ? bq  : (mat == 1) ? bk  : bv;

  const int lane = threadIdx.x & 31;
  const int nloc = lane & 15;
  const int half = lane >> 4;

  const __bf16* abase = X + st * 32 * D_MOD;
  const __bf16* bbase = WT + h * (DK_SZ * D_MOD);

  v8f acc[2][4] = {};
  for (int d = 0; d < D_MOD; d += 32) {
    v16bf a0 = load_a_frag(abase + d, D_MOD);
    v16bf a1 = load_a_frag(abase + 16 * D_MOD + d, D_MOD);
    v16bf b0 = load_b_frag(bbase + 0 * 16 * D_MOD + d, D_MOD);
    v16bf b1 = load_b_frag(bbase + 1 * 16 * D_MOD + d, D_MOD);
    v16bf b2 = load_b_frag(bbase + 2 * 16 * D_MOD + d, D_MOD);
    v16bf b3 = load_b_frag(bbase + 3 * 16 * D_MOD + d, D_MOD);
    acc[0][0] = wmma_bf16(a0, b0, acc[0][0]);
    acc[0][1] = wmma_bf16(a0, b1, acc[0][1]);
    acc[0][2] = wmma_bf16(a0, b2, acc[0][2]);
    acc[0][3] = wmma_bf16(a0, b3, acc[0][3]);
    acc[1][0] = wmma_bf16(a1, b0, acc[1][0]);
    acc[1][1] = wmma_bf16(a1, b1, acc[1][1]);
    acc[1][2] = wmma_bf16(a1, b2, acc[1][2]);
    acc[1][3] = wmma_bf16(a1, b3, acc[1][3]);
  }

  if (mat < 2) {
    __bf16* out = (mat == 0) ? qh : kh;                  // [H,S,DK]
#pragma unroll
    for (int j = 0; j < 4; ++j) {
      const float bias = bb[h * DK_SZ + j * 16 + nloc];
#pragma unroll
      for (int i = 0; i < 2; ++i) {
        __bf16* o = out + h * (S_LEN * DK_SZ)
                        + (st * 32 + i * 16) * DK_SZ + j * 16 + nloc;
#pragma unroll
        for (int r = 0; r < 8; ++r) {
          int M = r + half * 8;
          o[M * DK_SZ] = (__bf16)(acc[i][j][r] + bias);
        }
      }
    }
  } else {                                               // vT [H,DK,S]
#pragma unroll
    for (int j = 0; j < 4; ++j) {
      const float bias = bb[h * DK_SZ + j * 16 + nloc];
#pragma unroll
      for (int i = 0; i < 2; ++i) {
        __bf16* o = vT + h * (DK_SZ * S_LEN) + (j * 16 + nloc) * S_LEN
                       + st * 32 + i * 16 + half * 8;
#pragma unroll
        for (int r = 0; r < 8; ++r) o[r] = (__bf16)(acc[i][j][r] + bias);
      }
    }
  }
}

// ---------------------------------------------------------------------------
// Flash attention: one wave per (head, 16-query-row tile), online softmax
// over 32-key chunks. 8 WMMAs per chunk (4 scores + 4 P.V).
// ---------------------------------------------------------------------------
__global__ void __launch_bounds__(128)
attn_kernel(const __bf16* __restrict__ qh, const __bf16* __restrict__ kh,
            const __bf16* __restrict__ vT, __bf16* __restrict__ heads) {
  __shared__ __bf16 psmem[4][16 * 32];                   // P staging, 1KB/wave
  const int w    = threadIdx.x >> 5;
  const int lane = threadIdx.x & 31;
  const int nloc = lane & 15;
  const int half = lane >> 4;
  const int h    = blockIdx.y;
  const int s0   = blockIdx.x * 64 + w * 16;

  const __bf16* qbase = qh + h * (S_LEN * DK_SZ) + s0 * DK_SZ;
  v16bf qa0 = load_a_frag(qbase,      DK_SZ);            // dk 0..31
  v16bf qa1 = load_a_frag(qbase + 32, DK_SZ);            // dk 32..63

  v8f o0 = {}, o1 = {}, o2 = {}, o3 = {};
  float m[8], l[8];
#pragma unroll
  for (int r = 0; r < 8; ++r) { m[r] = -3.0e38f; l[r] = 0.0f; }

  __bf16* pl = &psmem[w][0];
  const __bf16* kbaseh = kh + h * (S_LEN * DK_SZ);
  const __bf16* vbaseh = vT + h * (DK_SZ * S_LEN);

  for (int key0 = 0; key0 < S_LEN; key0 += 32) {
    // ---- scores: two 16x16 tiles over this 32-key chunk ----
    v8f c0 = {}, c1 = {};
    {
      const __bf16* kb0 = kbaseh + key0 * DK_SZ;
      c0 = wmma_bf16(qa0, load_b_frag(kb0,      DK_SZ), c0);
      c0 = wmma_bf16(qa1, load_b_frag(kb0 + 32, DK_SZ), c0);
      const __bf16* kb1 = kbaseh + (key0 + 16) * DK_SZ;
      c1 = wmma_bf16(qa0, load_b_frag(kb1,      DK_SZ), c1);
      c1 = wmma_bf16(qa1, load_b_frag(kb1 + 32, DK_SZ), c1);
    }
    const float sc = 0.125f;                             // 1/sqrt(64)
#pragma unroll
    for (int r = 0; r < 8; ++r) {
      float x0 = c0[r] * sc, x1 = c1[r] * sc;
      // row max across the 16 lanes holding this row (stays in half-wave)
      float rm = fmaxf(x0, x1);
      rm = fmaxf(rm, __shfl_xor(rm, 1, 32));
      rm = fmaxf(rm, __shfl_xor(rm, 2, 32));
      rm = fmaxf(rm, __shfl_xor(rm, 4, 32));
      rm = fmaxf(rm, __shfl_xor(rm, 8, 32));
      float mn    = fmaxf(m[r], rm);
      float alpha = __expf(m[r] - mn);
      float p0    = __expf(x0 - mn);
      float p1    = __expf(x1 - mn);
      float rs = p0 + p1;
      rs += __shfl_xor(rs, 1, 32);
      rs += __shfl_xor(rs, 2, 32);
      rs += __shfl_xor(rs, 4, 32);
      rs += __shfl_xor(rs, 8, 32);
      l[r] = l[r] * alpha + rs;
      m[r] = mn;
      o0[r] *= alpha; o1[r] *= alpha; o2[r] *= alpha; o3[r] *= alpha;
      // stage P tile to LDS (C-layout -> memory row-major 16x32)
      int M = r + half * 8;
      pl[M * 32 + nloc]      = (__bf16)p0;
      pl[M * 32 + 16 + nloc] = (__bf16)p1;
    }
    asm volatile("s_wait_dscnt 0" ::: "memory");         // P visible in LDS
    v16bf pa = load_a_frag(pl, 32);                      // ds_load_b128 x2
    // ---- O += P @ V : 4 dv tiles, B frags from vT contiguous along keys ----
    o0 = wmma_bf16(pa, load_b_frag(vbaseh + (0 * 16) * S_LEN + key0, S_LEN), o0);
    o1 = wmma_bf16(pa, load_b_frag(vbaseh + (1 * 16) * S_LEN + key0, S_LEN), o1);
    o2 = wmma_bf16(pa, load_b_frag(vbaseh + (2 * 16) * S_LEN + key0, S_LEN), o2);
    o3 = wmma_bf16(pa, load_b_frag(vbaseh + (3 * 16) * S_LEN + key0, S_LEN), o3);
  }

  // epilogue: divide by l, write heads [S, H*DK] bf16
  __bf16* ob = heads + s0 * (NHEAD * DK_SZ) + h * DK_SZ + nloc;
#pragma unroll
  for (int r = 0; r < 8; ++r) {
    int M = r + half * 8;
    float inv = 1.0f / l[r];
    ob[M * (NHEAD * DK_SZ) + 0]  = (__bf16)(o0[r] * inv);
    ob[M * (NHEAD * DK_SZ) + 16] = (__bf16)(o1[r] * inv);
    ob[M * (NHEAD * DK_SZ) + 32] = (__bf16)(o2[r] * inv);
    ob[M * (NHEAD * DK_SZ) + 48] = (__bf16)(o3[r] * inv);
  }
}

// ---------------------------------------------------------------------------
// Output projection: out = heads @ Wo + bo  (f32 output)
// 32x64 block per wave, 8 WMMAs per K=32 chunk.
// ---------------------------------------------------------------------------
__global__ void __launch_bounds__(256)
outproj_kernel(const __bf16* __restrict__ heads, const __bf16* __restrict__ WoT,
               const float* __restrict__ bo, float* __restrict__ out) {
  const int gw = blockIdx.x * 8 + (threadIdx.x >> 5);    // 0..1023
  const int st = gw >> 4;                                // 64 row tiles (32 rows)
  const int dg = gw & 15;                                // 16 col groups (64 cols)
  const int lane = threadIdx.x & 31;
  const int nloc = lane & 15;
  const int half = lane >> 4;

  const __bf16* abase = heads + st * 32 * D_MOD;
  const __bf16* bbase = WoT + dg * 64 * D_MOD;

  v8f acc[2][4] = {};
  for (int kk = 0; kk < D_MOD; kk += 32) {
    v16bf a0 = load_a_frag(abase + kk, D_MOD);
    v16bf a1 = load_a_frag(abase + 16 * D_MOD + kk, D_MOD);
    v16bf b0 = load_b_frag(bbase + 0 * 16 * D_MOD + kk, D_MOD);
    v16bf b1 = load_b_frag(bbase + 1 * 16 * D_MOD + kk, D_MOD);
    v16bf b2 = load_b_frag(bbase + 2 * 16 * D_MOD + kk, D_MOD);
    v16bf b3 = load_b_frag(bbase + 3 * 16 * D_MOD + kk, D_MOD);
    acc[0][0] = wmma_bf16(a0, b0, acc[0][0]);
    acc[0][1] = wmma_bf16(a0, b1, acc[0][1]);
    acc[0][2] = wmma_bf16(a0, b2, acc[0][2]);
    acc[0][3] = wmma_bf16(a0, b3, acc[0][3]);
    acc[1][0] = wmma_bf16(a1, b0, acc[1][0]);
    acc[1][1] = wmma_bf16(a1, b1, acc[1][1]);
    acc[1][2] = wmma_bf16(a1, b2, acc[1][2]);
    acc[1][3] = wmma_bf16(a1, b3, acc[1][3]);
  }
#pragma unroll
  for (int j = 0; j < 4; ++j) {
    const float bias = bo[dg * 64 + j * 16 + nloc];
#pragma unroll
    for (int i = 0; i < 2; ++i) {
      float* ob = out + (st * 32 + i * 16) * D_MOD + dg * 64 + j * 16 + nloc;
#pragma unroll
      for (int r = 0; r < 8; ++r) ob[(r + half * 8) * D_MOD] = acc[i][j][r] + bias;
    }
  }
}

// ---------------------------------------------------------------------------
extern "C" void kernel_launch(void* const* d_in, const int* in_sizes, int n_in,
                              void* d_out, int out_size, void* d_ws, size_t ws_size,
                              hipStream_t stream) {
  (void)in_sizes; (void)n_in; (void)out_size; (void)ws_size;
  const float* Q  = (const float*)d_in[0];
  const float* K  = (const float*)d_in[1];
  const float* V  = (const float*)d_in[2];
  const float* Wq = (const float*)d_in[3];
  const float* bq = (const float*)d_in[4];
  const float* Wk = (const float*)d_in[5];
  const float* bk = (const float*)d_in[6];
  const float* Wv = (const float*)d_in[7];
  const float* bv = (const float*)d_in[8];
  const float* Wo = (const float*)d_in[9];
  const float* bo = (const float*)d_in[10];
  float* out = (float*)d_out;

  // workspace layout (bytes): 36 MB total
  char* ws = (char*)d_ws;
  const size_t MB = 1024 * 1024;
  __bf16* Qb    = (__bf16*)(ws + 0 * MB);    // S*D bf16   (4 MB)
  __bf16* Kb    = (__bf16*)(ws + 4 * MB);
  __bf16* Vb    = (__bf16*)(ws + 8 * MB);
  __bf16* WqT   = (__bf16*)(ws + 12 * MB);   // H*DK*D bf16 (2 MB)
  __bf16* WkT   = (__bf16*)(ws + 14 * MB);
  __bf16* WvT   = (__bf16*)(ws + 16 * MB);
  __bf16* WoT   = (__bf16*)(ws + 18 * MB);
  __bf16* qhp   = (__bf16*)(ws + 20 * MB);   // [H,S,DK] (4 MB)
  __bf16* khp   = (__bf16*)(ws + 24 * MB);
  __bf16* vTp   = (__bf16*)(ws + 28 * MB);   // [H,DK,S]
  __bf16* heads = (__bf16*)(ws + 32 * MB);   // [S,H*DK]

  const int nSD = S_LEN * D_MOD;             // 2,097,152
  cvt_f32_bf16<<<nSD / 256, 256, 0, stream>>>(Q, Qb, nSD);
  cvt_f32_bf16<<<nSD / 256, 256, 0, stream>>>(K, Kb, nSD);
  cvt_f32_bf16<<<nSD / 256, 256, 0, stream>>>(V, Vb, nSD);

  const int nW = NHEAD * D_MOD * DK_SZ;      // 1,048,576
  cvt_transpose_w<<<nW / 256, 256, 0, stream>>>(Wq, WqT);
  cvt_transpose_w<<<nW / 256, 256, 0, stream>>>(Wk, WkT);
  cvt_transpose_w<<<nW / 256, 256, 0, stream>>>(Wv, WvT);
  cvt_transpose_wo<<<(D_MOD * D_MOD) / 256, 256, 0, stream>>>(Wo, WoT);

  // 3 mats * 16 heads * 64 row-tiles = 3072 wave tiles / 8 waves per block
  proj_kernel<<<384, 256, 0, stream>>>(Qb, Kb, Vb, WqT, WkT, WvT,
                                       bq, bk, bv, qhp, khp, vTp);

  attn_kernel<<<dim3(S_LEN / 64, NHEAD), 128, 0, stream>>>(qhp, khp, vTp, heads);

  // 64 row-tiles * 16 col-groups = 1024 wave tiles / 8 waves per block
  outproj_kernel<<<128, 256, 0, stream>>>(heads, WoT, bo, out);
}
